// S3UMBlock_42734924595459
// MI455X (gfx1250) — matrix operand order repaired and checked by reading, not compiled, not run on hardware
//
#include <hip/hip_runtime.h>
#include <hip/hip_fp16.h>

typedef __attribute__((ext_vector_type(16))) _Float16 v16h;
typedef __attribute__((ext_vector_type(8)))  _Float16 v8h;
typedef __attribute__((ext_vector_type(8)))  float    v8f;
typedef __attribute__((address_space(3))) void* lds_vp;

#define DI static __device__ __forceinline__

static constexpr int Bsz = 2, Lsz = 2048, Dsz = 1024, Kcv = 4;
static constexpr int Mrows = Bsz * Lsz;   // 4096 token rows

DI float sigmf_(float x){ return 1.f/(1.f+__expf(-x)); }
DI float siluf_(float x){ return x*sigmf_(x); }
DI float softplusf_(float x){ return x>20.f ? x : __logf(1.f+__expf(x)); }

// ---------------- f32 -> f16 transpose-convert (Wt[n][k] = W[k][n]) ----------
__global__ __launch_bounds__(256) void k_cvt16t(const float* __restrict__ s,
                                                _Float16* __restrict__ d){
  __shared__ float t[32][33];
  int bx = blockIdx.x, by = blockIdx.y;          // k-tile, n-tile
  int tx = threadIdx.x & 31, ty = threadIdx.x >> 5;   // 32 x 8
  #pragma unroll
  for (int i=0;i<32;i+=8)
    t[ty+i][tx] = s[(size_t)(bx*32+ty+i)*Dsz + by*32+tx];   // [k][n], coalesced
  __syncthreads();
  #pragma unroll
  for (int i=0;i<32;i+=8)
    d[(size_t)(by*32+ty+i)*Dsz + bx*32+tx] = (_Float16)t[tx][ty+i];
}

// ---------------- LayerNorm + gate = silu(xn) ----------------
__global__ __launch_bounds__(256) void k_ln(const float* __restrict__ x,
                     const float* __restrict__ g, const float* __restrict__ b,
                     _Float16* __restrict__ xn16, float* __restrict__ gate){
  int row = blockIdx.x, tid = threadIdx.x;
  const float* xr = x + (size_t)row*Dsz;
  float v[4]; float s = 0.f;
  #pragma unroll
  for (int i=0;i<4;i++){ v[i] = xr[tid + 256*i]; s += v[i]; }
  __shared__ float red[256];
  red[tid] = s; __syncthreads();
  #pragma unroll
  for (int o=128;o>0;o>>=1){ if (tid<o) red[tid]+=red[tid+o]; __syncthreads(); }
  float mu = red[0]*(1.f/Dsz);
  __syncthreads();
  s = 0.f;
  #pragma unroll
  for (int i=0;i<4;i++){ float dd=v[i]-mu; s += dd*dd; }
  red[tid] = s; __syncthreads();
  #pragma unroll
  for (int o=128;o>0;o>>=1){ if (tid<o) red[tid]+=red[tid+o]; __syncthreads(); }
  float rstd = rsqrtf(red[0]*(1.f/Dsz)+1e-5f);
  #pragma unroll
  for (int i=0;i<4;i++){
    int c = tid + 256*i;
    float xn = (v[i]-mu)*rstd*g[c] + b[c];
    xn16[(size_t)row*Dsz + c] = (_Float16)xn;
    gate[(size_t)row*Dsz + c] = siluf_(xn);
  }
}

// ---------------- WMMA GEMM: C[M,Nd] = A[M,Kd](f16) * W[Kd,Nd] + bias --------
// W supplied TRANSPOSED as Wt[Nd][Kd] (f16).  epi: 0=none, 1=softplus, 2=+resid
#define GBM 64
#define GBN 128
#define GLA 40    // A tile pitch in halves (80B, 16B aligned)
#define GLB 40    // Bt tile pitch in halves (80B, 16B aligned)

union AFu { v16h v; v8h h8[2]; };

__global__ __launch_bounds__(256) void k_gemm(
    const _Float16* __restrict__ A, const _Float16* __restrict__ Wt,
    const float* __restrict__ bias, const float* __restrict__ resid,
    float* __restrict__ C, int Kd, int Nd, int epi)
{
  __shared__ _Float16 As[GBM*GLA];   // 64 rows x 32 k-halves
  __shared__ _Float16 Bt[GBN*GLB];   // 128 n-rows x 32 k-halves (transposed)
  int tid = threadIdx.x;
  int wave = tid>>5, lane = tid&31;
  int wm = wave>>2, wn = wave&3;           // 2 x 4 wave grid
  int rowBase = blockIdx.x*GBM;
  int colBase = blockIdx.y*GBN;

  v8f acc[2][2] = {};

  int ar = tid>>2, ac = (tid&3)*8;         // A: 8 halves/thread
  int br = tid>>1, bc = (tid&1)*16;        // Bt: 16 halves/thread
  const _Float16* gA = A  + (size_t)(rowBase+ar)*Kd + ac;
  const _Float16* gB = Wt + (size_t)(colBase+br)*Kd + bc;

  v8h ra  = *(const v8h*)gA;
  v8h rb0 = *(const v8h*)gB;
  v8h rb1 = *(const v8h*)(gB + 8);

  for (int kb=0; kb<Kd; kb+=32){
    *(v8h*)&As[ar*GLA + ac]     = ra;
    *(v8h*)&Bt[br*GLB + bc]     = rb0;
    *(v8h*)&Bt[br*GLB + bc + 8] = rb1;
    __syncthreads();

    if (kb+32 < Kd){                       // overlap next global loads with math
      ra  = *(const v8h*)(gA + kb + 32);
      rb0 = *(const v8h*)(gB + kb + 32);
      rb1 = *(const v8h*)(gB + kb + 40);
      if (kb+64 < Kd){
        __builtin_prefetch(gA + kb + 64, 0, 3);   // global_prefetch_b8
        __builtin_prefetch(gB + kb + 64, 0, 3);
      }
    }

    // A frag (ISA 7.12.2): lanes 0-15 row M=lane, slots 0..7->K0..7, 8..15->K16..23;
    //                      lanes 16-31 row M=lane-16, K8..15 / K24..31
    AFu af[2], bf[2];
    int klo = (lane<16) ? 0 : 8;
    #pragma unroll
    for (int fi=0; fi<2; fi++){
      int lrow = wm*32 + fi*16 + (lane&15);
      af[fi].h8[0] = *(const v8h*)&As[lrow*GLA + klo];
      af[fi].h8[1] = *(const v8h*)&As[lrow*GLA + 16 + klo];
    }
    // B frag: same K order as A frag (slots 0..7 -> K kadd..kadd+7, 8..15 -> K 16+kadd..), N = lane&15
    int kadd = (lane<16) ? 0 : 8;
    #pragma unroll
    for (int fj=0; fj<2; fj++){
      int ncol = wn*32 + fj*16 + (lane&15);
      bf[fj].h8[0] = *(const v8h*)&Bt[ncol*GLB + kadd];
      bf[fj].h8[1] = *(const v8h*)&Bt[ncol*GLB + kadd + 16];
    }
    #pragma unroll
    for (int fi=0; fi<2; fi++)
      #pragma unroll
      for (int fj=0; fj<2; fj++)
        acc[fi][fj] = __builtin_amdgcn_wmma_f32_16x16x32_f16(
            false, af[fi].v, false, bf[fj].v, (short)0, acc[fi][fj], false, false);
    __syncthreads();
  }

  // Epilogue: C/D layout — VGPR r: M = r (+8 for lanes 16-31), N = lane&15
  #pragma unroll
  for (int fi=0; fi<2; fi++){
    int mb = rowBase + wm*32 + fi*16 + ((lane<16)?0:8);
    #pragma unroll
    for (int fj=0; fj<2; fj++){
      int n = colBase + wn*32 + fj*16 + (lane&15);
      float bv = bias[n];
      #pragma unroll
      for (int r=0; r<8; r++){
        float val = acc[fi][fj][r] + bv;
        if (epi==1) val = softplusf_(val);
        size_t idx = (size_t)(mb+r)*Nd + n;
        if (epi==2) val += resid[idx];
        C[idx] = val;
      }
    }
  }
}

// ---------------- depthwise causal conv (K=4) + SiLU ----------------
__global__ void k_conv(const float* __restrict__ lp, const float* __restrict__ cw,
                       const float* __restrict__ cb, float* __restrict__ left,
                       _Float16* __restrict__ left16){
  size_t i = (size_t)blockIdx.x*256 + threadIdx.x;
  int d = (int)(i % Dsz);
  size_t bl = i / Dsz;
  int l = (int)(bl % Lsz);
  float acc = cb[d];
  #pragma unroll
  for (int k=0;k<Kcv;k++){
    int ll = l - 3 + k;
    if (ll >= 0) acc += cw[d*Kcv + k] * lp[(bl - 3 + k)*Dsz + d];
  }
  float s = siluf_(acc);
  left[i] = s; left16[i] = (_Float16)s;
}

// ---------------- bc = left @ w_bc  (N = 32, lane-per-output) ----------------
__global__ __launch_bounds__(256) void k_bc(const float* __restrict__ left,
                     const float* __restrict__ wbc, float* __restrict__ bc){
  int wave = threadIdx.x >> 5, lane = threadIdx.x & 31;
  int row = blockIdx.x*8 + wave;
  const float* lr = left + (size_t)row*Dsz;
  float acc = 0.f;
  for (int k=0;k<Dsz;k++) acc += lr[k] * wbc[k*32 + lane];   // lr[k] wave-uniform
  bc[(size_t)row*32 + lane] = acc;
}

// ---------------- sequential selective scan, fused gate + f16 convert --------
// B/C row staged into LDS via async global->LDS DMA (ASYNCcnt path, ISA 15.18.3)
__global__ __launch_bounds__(256) void k_scan(const float* __restrict__ dt,
                     const float* __restrict__ bc, const float* __restrict__ left,
                     const float* __restrict__ gate, const float* __restrict__ A_log,
                     const float* __restrict__ D_ssm, _Float16* __restrict__ ymul16){
  int d = blockIdx.x*256 + threadIdx.x;
  int b = blockIdx.y;
  float Ad[16];
  #pragma unroll
  for (int n=0;n<16;n++) Ad[n] = -__expf(A_log[d*16+n]);
  float Dv = D_ssm[d];
  float h[16];
  #pragma unroll
  for (int n=0;n<16;n++) h[n] = 0.f;
  __shared__ float buf[2][32];
  const size_t rbase = (size_t)b*Lsz;
  for (int t=0; t<Lsz; t++){
    if (threadIdx.x < 32) {   // wave-uniform: wave 0 only
      // per-lane async DMA: 4B from bc row -> LDS, tracked by ASYNCcnt
      unsigned loff = (unsigned)(size_t)(lds_vp)&buf[t&1][threadIdx.x];
      unsigned long long gaddr =
          (unsigned long long)(const void*)&bc[(rbase+t)*32 + threadIdx.x];
      asm volatile("global_load_async_to_lds_b32 %0, %1, off"
                   :: "v"(loff), "v"(gaddr) : "memory");
      asm volatile("s_wait_asynccnt 0x0" ::: "memory");
    }
    __syncthreads();   // double buffer: buf[(t+1)&1] last read before sync(t)
    size_t idx = (rbase+t)*Dsz + d;
    float dtv = dt[idx], lv = left[idx];
    const float* Bt = buf[t&1];
    const float* Ct = buf[t&1] + 16;
    float y = 0.f;
    #pragma unroll
    for (int n=0;n<16;n++){
      float dA = __expf(dtv*Ad[n]);
      h[n] = dA*h[n] + dtv*Bt[n]*lv;
      y += h[n]*Ct[n];
    }
    float ym = (y + lv*Dv) * gate[idx];
    ymul16[idx] = (_Float16)ym;
  }
}

extern "C" void kernel_launch(void* const* d_in, const int* in_sizes, int n_in,
                              void* d_out, int out_size, void* d_ws, size_t ws_size,
                              hipStream_t stream) {
  const float* x      = (const float*)d_in[0];
  const float* ln_g   = (const float*)d_in[1];
  const float* ln_b   = (const float*)d_in[2];
  const float* w_in   = (const float*)d_in[3];
  const float* b_in   = (const float*)d_in[4];
  const float* conv_w = (const float*)d_in[5];
  const float* conv_b = (const float*)d_in[6];
  const float* A_log  = (const float*)d_in[7];
  const float* D_ssm  = (const float*)d_in[8];
  const float* w_dt   = (const float*)d_in[9];
  const float* b_dt   = (const float*)d_in[10];
  const float* w_bc   = (const float*)d_in[11];
  const float* w_out  = (const float*)d_in[12];
  const float* b_out  = (const float*)d_in[13];
  float* outp = (float*)d_out;

  char* p = (char*)d_ws;
  auto carve = [&](size_t bytes)->char* {
    char* r = p; p += (bytes + 255) & ~(size_t)255; return r;
  };
  const size_t MD = (size_t)Mrows*Dsz;
  _Float16* xn16    = (_Float16*)carve(MD*2);
  float*    gate    = (float*)   carve(MD*4);
  float*    leftp   = (float*)   carve(MD*4);
  float*    left    = (float*)   carve(MD*4);
  _Float16* left16  = (_Float16*)carve(MD*2);
  float*    dtb     = (float*)   carve(MD*4);
  float*    bcb     = (float*)   carve((size_t)Mrows*32*4);
  _Float16* ymul16  = (_Float16*)carve(MD*2);
  _Float16* w_in16t = (_Float16*)carve((size_t)Dsz*Dsz*2);
  _Float16* w_dt16t = (_Float16*)carve((size_t)Dsz*Dsz*2);
  _Float16* w_out16t= (_Float16*)carve((size_t)Dsz*Dsz*2);

  dim3 gt(Dsz/32, Dsz/32);
  k_cvt16t<<<gt, 256, 0, stream>>>(w_in,  w_in16t);
  k_cvt16t<<<gt, 256, 0, stream>>>(w_dt,  w_dt16t);
  k_cvt16t<<<gt, 256, 0, stream>>>(w_out, w_out16t);

  k_ln<<<Mrows, 256, 0, stream>>>(x, ln_g, ln_b, xn16, gate);

  dim3 gg(Mrows/GBM, Dsz/GBN);   // 64 x 8 blocks
  k_gemm<<<gg, 256, 0, stream>>>(xn16, w_in16t, b_in, nullptr, leftp, Dsz, Dsz, 0);

  k_conv<<<(unsigned)(MD/256), 256, 0, stream>>>(leftp, conv_w, conv_b, left, left16);

  k_gemm<<<gg, 256, 0, stream>>>(left16, w_dt16t, b_dt, nullptr, dtb, Dsz, Dsz, 1);

  k_bc<<<Mrows/8, 256, 0, stream>>>(left, w_bc, bcb);

  dim3 gs(Dsz/256, Bsz);
  k_scan<<<gs, 256, 0, stream>>>(dtb, bcb, left, gate, A_log, D_ssm, ymul16);

  k_gemm<<<gg, 256, 0, stream>>>(ymul16, w_out16t, b_out, x, outp, Dsz, Dsz, 2);
}
